// DeformConv3D_63058709840070
// MI455X (gfx1250) — compile-verified, matrix operand order checked by reading, not run on hardware
//
#include <hip/hip_runtime.h>
#include <hip/hip_bf16.h>

// ---------------------------------------------------------------------------
// DeformConv3D for MI455X (gfx1250, wave32).
// B=2, CIN=64, COUT=128, D=H=W=64.  P = B*D*H*W = 524288 points.
// Fused: offset conv (VALU) -> trilinear gather (vmem, L2-resident) ->
//        main 1x1x1 conv as bf16 WMMA GEMM ([P x 64] * [64 x 128]).
// CDNA5 specifics: v_wmma_f32_16x16x32_bf16, global_load_async_to_lds_b128
// weight staging (ASYNCcnt), non-temporal output stores to keep x in L2.
// Phase 2 keeps dx fragments resident in VGPRs across all 8 cout tiles and
// shares each weight fragment across both point tiles (40 ds_load_b128/wave
// instead of 128) to cut register pressure and raise occupancy.
// ---------------------------------------------------------------------------

#define CIN   64
#define COUT  128
#define DIM   64
#define DHW   262144          // 64*64*64 = 2^18
#define PTS_PER_BLOCK 256
#define ROWPAD 72             // 64 bf16 + 8 pad (16B-aligned rows, bank stagger)

typedef __attribute__((ext_vector_type(16))) __bf16 v16bf;
typedef __attribute__((ext_vector_type(8)))  __bf16 v8bf;
typedef __attribute__((ext_vector_type(8)))  float  v8f;

// float -> bf16 (round to nearest even), bit-exact, no scalar __bf16 casts
static __device__ __forceinline__ unsigned short f2bf(float f) {
    unsigned u = __float_as_uint(f);
    u += 0x7FFFu + ((u >> 16) & 1u);
    return (unsigned short)(u >> 16);
}

static __device__ __forceinline__ float clampf(float v, float lo, float hi) {
    return fminf(fmaxf(v, lo), hi);
}

// Build a 16-element bf16 WMMA fragment from an LDS row.
// 16-bit A/B layout: this lane-half holds K = kc*32 + kbase + {0..7} in regs 0-3
// and K = kc*32 + 16 + kbase + {0..7} in regs 4-7 (kbase = 0 or 8).
static __device__ __forceinline__ v16bf load_frag(const unsigned short* row,
                                                  int kc, int kbase) {
    const v8bf lo = *(const v8bf*)(row + kc * 32 + kbase);
    const v8bf hi = *(const v8bf*)(row + kc * 32 + 16 + kbase);
    return __builtin_shufflevector(lo, hi,
                                   0, 1, 2, 3, 4, 5, 6, 7,
                                   8, 9, 10, 11, 12, 13, 14, 15);
}

__global__ __launch_bounds__(256)
void DeformConv3D_63058709840070_kernel(const float* __restrict__ x,
                                        const float* __restrict__ w_off,
                                        const float* __restrict__ b_off,
                                        const float* __restrict__ w_conv,
                                        const float* __restrict__ b_conv,
                                        float* __restrict__ out) {
    __shared__ __align__(16) unsigned short s_dx[PTS_PER_BLOCK * ROWPAD]; // bf16 feats / fp32 stage
    __shared__ __align__(16) unsigned short s_wc[COUT * ROWPAD];          // bf16 conv weights
    __shared__ float s_woff[3 * CIN + 4];                                 // w_off rows + b_off
    __shared__ float s_bconv[COUT];

    const int tid = threadIdx.x;

    // ---- Phase 0a: async-DMA w_conv (fp32, 32KB) into LDS staging --------
    // global_load_async_to_lds_* bypasses VGPRs and is tracked by ASYNCcnt.
    {
        const unsigned ldsBase = (unsigned)(uintptr_t)(void*)s_dx; // low bits = LDS offset
        #pragma unroll
        for (int i = tid; i < (COUT * CIN) / 4; i += PTS_PER_BLOCK) {
            const unsigned lA = ldsBase + (unsigned)i * 16u;
            const unsigned gA = (unsigned)i * 16u;
            asm volatile("global_load_async_to_lds_b128 %0, %1, %2"
                         :: "v"(lA), "v"(gA), "s"(w_conv) : "memory");
        }
        asm volatile("s_wait_asynccnt 0x0" ::: "memory");
    }
    if (tid < 3 * CIN)  s_woff[tid] = w_off[tid];
    if (tid >= 3 * CIN && tid < 3 * CIN + 3) s_woff[tid] = b_off[tid - 3 * CIN];
    if (tid < COUT)     s_bconv[tid] = b_conv[tid];
    __syncthreads();

    // ---- Phase 0b: convert staged fp32 weights -> bf16 (padded rows) -----
    {
        const float* stage = (const float*)(const void*)s_dx;
        #pragma unroll
        for (int i = tid; i < COUT * CIN; i += PTS_PER_BLOCK) {
            s_wc[(i >> 6) * ROWPAD + (i & 63)] = f2bf(stage[i]);
        }
    }
    __syncthreads();   // staging buffer about to be reused for dx

    // ---- Phase 1: per-lane offset conv + trilinear sampling --------------
    const int p0   = blockIdx.x * PTS_PER_BLOCK;     // first global point
    const int p    = p0 + tid;                       // this lane's point
    const int bIdx = p >> 18;                        // batch (DHW = 2^18)
    const int s    = p & (DHW - 1);                  // spatial index
    const int wI   = s & 63;
    const int yI   = (s >> 6) & 63;
    const int zI   = (s >> 12) & 63;

    const float* xb = x + ((unsigned)bIdx << 24);    // batch base (b*CIN*DHW)

    // offset conv: 64 coalesced loads (consecutive w across lanes)
    float ox = s_woff[192], oy = s_woff[193], oz = s_woff[194];
    {
        const float* xc = xb + s;
        #pragma unroll 8
        for (int c = 0; c < CIN; ++c) {
            const float v = xc[(unsigned)c << 18];
            ox = fmaf(s_woff[c],       v, ox);
            oy = fmaf(s_woff[64 + c],  v, oy);
            oz = fmaf(s_woff[128 + c], v, oz);
        }
    }

    // base grid (linspace(-1,1,64)) + offset, unnormalize (align_corners=False),
    // border clamp
    const float inv63x2 = 2.0f / 63.0f;
    const float gx = fmaf((float)wI, inv63x2, -1.0f) + ox;
    const float gy = fmaf((float)yI, inv63x2, -1.0f) + oy;
    const float gz = fmaf((float)zI, inv63x2, -1.0f) + oz;
    const float ix = clampf(fmaf(gx + 1.0f, 32.0f, -0.5f), 0.0f, 63.0f);
    const float iy = clampf(fmaf(gy + 1.0f, 32.0f, -0.5f), 0.0f, 63.0f);
    const float iz = clampf(fmaf(gz + 1.0f, 32.0f, -0.5f), 0.0f, 63.0f);

    const float fx0 = floorf(ix), fy0 = floorf(iy), fz0 = floorf(iz);
    const int ix0 = (int)fx0, iy0 = (int)fy0, iz0 = (int)fz0;
    const int ix1 = min(ix0 + 1, 63), iy1 = min(iy0 + 1, 63), iz1 = min(iz0 + 1, 63);
    const float fx = ix - fx0, fy = iy - fy0, fz = iz - fz0;
    const float wx0 = 1.0f - fx, wy0 = 1.0f - fy, wz0 = 1.0f - fz;

    const float w000 = wz0 * wy0 * wx0, w001 = wz0 * wy0 * fx;
    const float w010 = wz0 * fy  * wx0, w011 = wz0 * fy  * fx;
    const float w100 = fz  * wy0 * wx0, w101 = fz  * wy0 * fx;
    const float w110 = fz  * fy  * wx0, w111 = fz  * fy  * fx;

    const int o000 = iz0 * 4096 + iy0 * 64 + ix0;
    const int o001 = iz0 * 4096 + iy0 * 64 + ix1;
    const int o010 = iz0 * 4096 + iy1 * 64 + ix0;
    const int o011 = iz0 * 4096 + iy1 * 64 + ix1;
    const int o100 = iz1 * 4096 + iy0 * 64 + ix0;
    const int o101 = iz1 * 4096 + iy0 * 64 + ix1;
    const int o110 = iz1 * 4096 + iy1 * 64 + ix0;
    const int o111 = iz1 * 4096 + iy1 * 64 + ix1;

    // gather 64 channels (L2-resident: whole x fits in 192MB L2), pack bf16 pairs
    unsigned* dx32 = (unsigned*)s_dx + (unsigned)tid * (ROWPAD / 2);
    #pragma unroll 4
    for (int c = 0; c < CIN; c += 2) {
        const float* pc0 = xb + ((unsigned)c << 18);
        const float* pc1 = pc0 + DHW;
        float a0, a1;
        a0 = w000 * pc0[o000];            a1 = w000 * pc1[o000];
        a0 = fmaf(w001, pc0[o001], a0);   a1 = fmaf(w001, pc1[o001], a1);
        a0 = fmaf(w010, pc0[o010], a0);   a1 = fmaf(w010, pc1[o010], a1);
        a0 = fmaf(w011, pc0[o011], a0);   a1 = fmaf(w011, pc1[o011], a1);
        a0 = fmaf(w100, pc0[o100], a0);   a1 = fmaf(w100, pc1[o100], a1);
        a0 = fmaf(w101, pc0[o101], a0);   a1 = fmaf(w101, pc1[o101], a1);
        a0 = fmaf(w110, pc0[o110], a0);   a1 = fmaf(w110, pc1[o110], a1);
        a0 = fmaf(w111, pc0[o111], a0);   a1 = fmaf(w111, pc1[o111], a1);
        dx32[c >> 1] = (unsigned)f2bf(a0) | ((unsigned)f2bf(a1) << 16);
    }
    __syncthreads();

    // ---- Phase 2: main conv as bf16 WMMA GEMM ----------------------------
    // D[cout, point] = W[cout, k] * dx^T[k, point]; 2 K-chunks of 32.
    // dx fragments stay resident in VGPRs across all cout tiles; each weight
    // fragment is loaded once and shared by both point tiles.
    const int wid   = tid >> 5;
    const int lane  = tid & 31;
    const int lo16  = lane & 15;
    const int hi    = lane >> 4;      // lane-half
    const int kbase = hi * 8;

    const int bBlk = p0 >> 18;                 // whole block shares batch
    const int s0   = p0 & (DHW - 1);           // block spatial base
    // 32-bit element offsets: total out = 67M elems (268MB) < 2^31 bytes.
    const unsigned outBase = (unsigned)bBlk * (unsigned)(COUT * DHW) + (unsigned)s0;

    const int tile0 = wid * 2;                 // this wave's two 16-point tiles
    const unsigned short* drow0 = &s_dx[(unsigned)(tile0 * 16 + lo16) * ROWPAD];
    const unsigned short* drow1 = drow0 + 16 * ROWPAD;

    const v16bf b00 = load_frag(drow0, 0, kbase);   // tile0, K 0..31
    const v16bf b01 = load_frag(drow0, 1, kbase);   // tile0, K 32..63
    const v16bf b10 = load_frag(drow1, 0, kbase);   // tile1, K 0..31
    const v16bf b11 = load_frag(drow1, 1, kbase);   // tile1, K 32..63

    const unsigned oPt0 = outBase + (unsigned)(tile0 * 16 + lo16);
    const unsigned oPt1 = oPt0 + 16u;

    for (int ct = 0; ct < 8; ++ct) {           // cout tiles of 16
        const unsigned short* wrow = &s_wc[(unsigned)(ct * 16 + lo16) * ROWPAD];
        const v16bf a0 = load_frag(wrow, 0, kbase);
        const v16bf a1 = load_frag(wrow, 1, kbase);

        v8f acc0 = {}, acc1 = {};
        acc0 = __builtin_amdgcn_wmma_f32_16x16x32_bf16(false, a0, false, b00,
                                                       (short)0, acc0, false, false);
        acc1 = __builtin_amdgcn_wmma_f32_16x16x32_bf16(false, a0, false, b10,
                                                       (short)0, acc1, false, false);
        acc0 = __builtin_amdgcn_wmma_f32_16x16x32_bf16(false, a1, false, b01,
                                                       (short)0, acc0, false, false);
        acc1 = __builtin_amdgcn_wmma_f32_16x16x32_bf16(false, a1, false, b11,
                                                       (short)0, acc1, false, false);

        // C layout: lane holds point N=lo16; reg r holds cout M = r + hi*8.
        const int coutBase = ct * 16 + hi * 8;
        const unsigned oCt0 = oPt0 + (unsigned)coutBase * (unsigned)DHW;
        const unsigned oCt1 = oPt1 + (unsigned)coutBase * (unsigned)DHW;
        #pragma unroll
        for (int r = 0; r < 8; ++r) {
            const float bias = s_bconv[coutBase + r];
            // Non-temporal: out is a 268MB stream; keep L2 for x gathers.
            __builtin_nontemporal_store(acc0[r] + bias,
                                        &out[oCt0 + (unsigned)r * (unsigned)DHW]);
            __builtin_nontemporal_store(acc1[r] + bias,
                                        &out[oCt1 + (unsigned)r * (unsigned)DHW]);
        }
    }
}

extern "C" void kernel_launch(void* const* d_in, const int* in_sizes, int n_in,
                              void* d_out, int out_size, void* d_ws, size_t ws_size,
                              hipStream_t stream) {
    (void)in_sizes; (void)n_in; (void)d_ws; (void)ws_size; (void)out_size;
    const float* x      = (const float*)d_in[0];
    const float* w_off  = (const float*)d_in[1];
    const float* b_off  = (const float*)d_in[2];
    const float* w_conv = (const float*)d_in[3];
    const float* b_conv = (const float*)d_in[4];
    float* out = (float*)d_out;

    const int P = 2 * DHW;                      // B * D * H * W
    dim3 grid(P / PTS_PER_BLOCK);               // 2048 blocks
    dim3 block(PTS_PER_BLOCK);                  // 8 wave32s
    DeformConv3D_63058709840070_kernel<<<grid, block, 0, stream>>>(
        x, w_off, b_off, w_conv, b_conv, out);
}
